// GraphConvolution_2594160247373
// MI455X (gfx1250) — matrix operand order repaired and checked by reading, not compiled, not run on hardware
//
#include <hip/hip_runtime.h>

// GCN layer fused kernel for MI455X (gfx1250, wave32, WMMA).
// out = relu((A @ X) @ W + bias)
//   A: [N,N] fp32, X: [N,F] fp32, W: [F,U] fp32, bias: [U], out: [N,U] fp32
// N=20000, F=U=128.
//
// Precision strategy: A is pre-scaled by 2^15 (exact) and converted to f16 in
// registers so v_wmma_f32_16x16x32_f16 (XDL) can run at ~8x the fp32-WMMA
// rate; accumulation stays fp32 and is unscaled by 2^-15 once (exact).

typedef __attribute__((ext_vector_type(16))) _Float16 v16h;
typedef __attribute__((ext_vector_type(8)))  float    v8f;

union Frag16 {                 // one WMMA 16-bit operand fragment (8 VGPRs)
    v16h v;
    unsigned int u[8];
    _Float16 h[16];
};

#define NROWS 20000
#define FDIM  128
#define UDIM  128
#define KC    160              // K-chunk (divides 20000; 5 WMMA K-steps)
#define FPITCH 132             // dword pitch for staged X pairs (bank-safe)
#define SPITCH 136             // f16 pitch for supports tile (16B-aligned rows)
#define ASCALE 32768.0f        // 2^15, makes A values f16-normal
#define AUNSCALE (1.0f / 32768.0f)

__device__ __forceinline__ unsigned int pack_h2(float a, float b) {
    union { _Float16 h[2]; unsigned int u; } t;
    t.h[0] = (_Float16)a;      // K even  -> bits [15:0]
    t.h[1] = (_Float16)b;      // K odd   -> bits [31:16]
    return t.u;
}

__global__ __launch_bounds__(256) void
gcn_fused_wmma(const float* __restrict__ X,
               const float* __restrict__ A,
               const float* __restrict__ W,
               const float* __restrict__ bias,
               float* __restrict__ out) {
    __shared__ union {
        unsigned int feat[(KC / 2) * FPITCH];   // staged X chunk, packed f16 pairs
        _Float16     sup[128 * SPITCH];         // supports tile (reused after GEMM1)
    } lds;

    const int tid   = threadIdx.x;
    const int lane  = tid & 31;
    const int wave  = tid >> 5;          // 0..7  -> 16-row strip
    const int n16   = lane & 15;         // column within 16-wide tile
    const int h     = lane >> 4;         // lane half: selects K sub-range
    const int rowBase = blockIdx.x * 128;

    // Row of A this lane supplies to its A-fragments (clamped for tail block;
    // results for clamped rows are never stored).
    int mRow = rowBase + wave * 16 + n16;
    if (mRow > NROWS - 1) mRow = NROWS - 1;
    const long long aRow = (long long)mRow * NROWS;

    v8f acc[8];
    #pragma unroll
    for (int i = 0; i < 8; ++i)
        #pragma unroll
        for (int j = 0; j < 8; ++j) acc[i][j] = 0.0f;

    // ---------------- GEMM1: supports = A @ X (f16 WMMA, f32 accum) ----------
    for (int kc = 0; kc < NROWS; kc += KC) {
        __syncthreads();
        // Stage X[kc..kc+KC) as packed f16 pairs {X[2rp], X[2rp+1]} per dword,
        // matching the 16-bit B-fragment dword layout. Coalesced global reads.
        for (int p = tid; p < (KC / 2) * FDIM; p += 256) {
            const int rp  = p >> 7;        // FDIM == 128
            const int col = p & 127;
            const float f0 = X[(kc + 2 * rp    ) * FDIM + col];
            const float f1 = X[(kc + 2 * rp + 1) * FDIM + col];
            lds.feat[rp * FPITCH + col] = pack_h2(f0, f1);
        }
        __syncthreads();

        if (kc + KC < NROWS)               // pull next A chunk toward L2
            __builtin_prefetch(&A[aRow + kc + KC], 0, 1);

        #pragma unroll
        for (int ks = 0; ks < KC; ks += 32) {
            // A fragment: lane holds K = {16h..16h+15} of row mRow.
            Frag16 af;
            {
                const float4* ap0 = (const float4*)(A + aRow + kc + ks + 16 * h);
                const float4* ap1 = (const float4*)(A + aRow + kc + ks + 8 + 16 * h);
                const float4 a0 = ap0[0], a1 = ap0[1];
                const float4 a2 = ap1[0], a3 = ap1[1];
                af.u[0] = pack_h2(a0.x * ASCALE, a0.y * ASCALE);
                af.u[1] = pack_h2(a0.z * ASCALE, a0.w * ASCALE);
                af.u[2] = pack_h2(a1.x * ASCALE, a1.y * ASCALE);
                af.u[3] = pack_h2(a1.z * ASCALE, a1.w * ASCALE);
                af.u[4] = pack_h2(a2.x * ASCALE, a2.y * ASCALE);
                af.u[5] = pack_h2(a2.z * ASCALE, a2.w * ASCALE);
                af.u[6] = pack_h2(a3.x * ASCALE, a3.y * ASCALE);
                af.u[7] = pack_h2(a3.z * ASCALE, a3.w * ASCALE);
            }
            const int rbase = (ks >> 1) + 8 * h;   // pair-row in staged chunk
            #pragma unroll
            for (int nt = 0; nt < 8; ++nt) {
                Frag16 bf;
                #pragma unroll
                for (int j = 0; j < 8; ++j)
                    bf.u[j] = lds.feat[(rbase + j) * FPITCH + nt * 16 + n16];
                acc[nt] = __builtin_amdgcn_wmma_f32_16x16x32_f16(
                    false, af.v, false, bf.v, (short)0, acc[nt], false, false);
            }
        }
    }

    // Unscale (exact) and park supports tile in LDS as f16.
    __syncthreads();
    #pragma unroll
    for (int nt = 0; nt < 8; ++nt)
        #pragma unroll
        for (int r = 0; r < 8; ++r) {
            const int row = wave * 16 + r + 8 * h;   // C layout: VGPR r -> M=r(+8h)
            const int col = nt * 16 + n16;
            lds.sup[row * SPITCH + col] = (_Float16)(acc[nt][r] * AUNSCALE);
        }
    __syncthreads();

    // ---------------- GEMM2: out = relu(supports @ W + bias) -----------------
    v8f acc2[8];
    #pragma unroll
    for (int i = 0; i < 8; ++i)
        #pragma unroll
        for (int j = 0; j < 8; ++j) acc2[i][j] = 0.0f;

    const int m2 = wave * 16 + n16;      // supports row feeding this lane's A-frag
    #pragma unroll
    for (int k2 = 0; k2 < FDIM; k2 += 32) {
        Frag16 af;
        {
            const uint4* sp0 = (const uint4*)&lds.sup[m2 * SPITCH + k2 + 16 * h];
            const uint4* sp1 = (const uint4*)&lds.sup[m2 * SPITCH + k2 + 8 + 16 * h];
            const uint4 s0 = sp0[0], s1 = sp1[0];
            af.u[0] = s0.x; af.u[1] = s0.y; af.u[2] = s0.z; af.u[3] = s0.w;
            af.u[4] = s1.x; af.u[5] = s1.y; af.u[6] = s1.z; af.u[7] = s1.w;
        }
        #pragma unroll
        for (int nt = 0; nt < 8; ++nt) {
            Frag16 bf;
            #pragma unroll
            for (int j = 0; j < 8; ++j) {
                const int f = k2 + 16 * h + 2 * j;
                bf.u[j] = pack_h2(W[f * UDIM + nt * 16 + n16],
                                  W[(f + 1) * UDIM + nt * 16 + n16]);
            }
            acc2[nt] = __builtin_amdgcn_wmma_f32_16x16x32_f16(
                false, af.v, false, bf.v, (short)0, acc2[nt], false, false);
        }
    }

    // Bias + ReLU + guarded store.
    #pragma unroll
    for (int nt = 0; nt < 8; ++nt) {
        const int col = nt * 16 + n16;
        const float b = bias[col];
        #pragma unroll
        for (int r = 0; r < 8; ++r) {
            const int row = rowBase + wave * 16 + r + 8 * h;
            if (row < NROWS) {
                float v = acc2[nt][r] + b;
                out[row * UDIM + col] = v > 0.0f ? v : 0.0f;
            }
        }
    }
}

extern "C" void kernel_launch(void* const* d_in, const int* in_sizes, int n_in,
                              void* d_out, int out_size, void* d_ws, size_t ws_size,
                              hipStream_t stream) {
    const float* X    = (const float*)d_in[0];   // features [N,F]
    const float* A    = (const float*)d_in[1];   // adjacency [N,N]
    const float* W    = (const float*)d_in[2];   // kernel [F,U]
    const float* bias = (const float*)d_in[3];   // bias [U]
    float* out = (float*)d_out;                  // [N,U] fp32

    const int nBlocks = (NROWS + 127) / 128;     // 157
    gcn_fused_wmma<<<nBlocks, 256, 0, stream>>>(X, A, W, bias, out);
}
